// Seq2seq_26388279067014
// MI455X (gfx1250) — compile-verified, matrix operand order checked by reading, not run on hardware
//
#include <hip/hip_runtime.h>
#include <hip/hip_bf16.h>

// ---------------------------------------------------------------------------
// CDNA5 (gfx1250) WMMA helpers: v_wmma_f32_16x16x32_f16, wave32.
// A fragment (16x32 f16, MxK): lane = (row&15) + 16*kh; holds K = kh*8..+7
//   in elements 0..7 and K = 16+kh*8..+7 in elements 8..15.
// B fragment (32x16 f16, KxN): lane n holds column n of B == row n of the
//   row-major weight matrix W[N][K]; same K chunking as A.
// C/D fragment (16x16 f32): element i of lane L -> row i+8*(L>>4), col L&15.
// ---------------------------------------------------------------------------
typedef __attribute__((ext_vector_type(16))) _Float16 v16h;
typedef __attribute__((ext_vector_type(8)))  float    v8f;

__device__ inline v8f zero8() {
    v8f z;
#pragma unroll
    for (int i = 0; i < 8; ++i) z[i] = 0.0f;
    return z;
}

__device__ inline v8f wmma32(v16h a, v16h b, v8f c) {
    return __builtin_amdgcn_wmma_f32_16x16x32_f16(
        /*neg_a=*/false, a, /*neg_b=*/false, b,
        /*c_mod=*/(short)0, c, /*reuse_a=*/false, /*reuse_b=*/false);
}

// Build a 16-half fragment slice from a row-major f32 row pointer.
// p must point at (row_base + k0 + kh*8); reads p[0..7] and p[16..23].
__device__ inline v16h frag_f32(const float* p) {
    v16h f;
#pragma unroll
    for (int i = 0; i < 8; ++i) {
        f[i]     = (_Float16)p[i];
        f[8 + i] = (_Float16)p[16 + i];
    }
    return f;
}

__device__ inline v16h frag_f16(const _Float16* p) {
    v16h f;
#pragma unroll
    for (int i = 0; i < 8; ++i) {
        f[i]     = p[i];
        f[8 + i] = p[16 + i];
    }
    return f;
}

__device__ inline float sigmoidf_(float x) { return 1.0f / (1.0f + __expf(-x)); }

// Problem constants
#define BB 16
#define SS 256
#define TT 48
#define EE 256
#define HH 512
#define VV 32000

// ---------------------------------------------------------------------------
// Tiny helper kernels
// ---------------------------------------------------------------------------
__global__ void build_idx_kernel(const int* __restrict__ src,
                                 const int* __restrict__ tgt,
                                 int* __restrict__ idx_src,
                                 int* __restrict__ idx_tgt) {
    int m = blockIdx.x * 256 + threadIdx.x;
    if (m < SS * BB) idx_src[m] = src[(m & 15) * SS + (m >> 4)];   // row m=(t*16+b)
    if (m < 47 * BB) idx_tgt[m] = tgt[(m & 15) * TT + (m >> 4)];
}

__global__ void dec_embed_kernel(const int* __restrict__ tgt,
                                 const float* __restrict__ demb,
                                 float* __restrict__ dec_e,
                                 float* __restrict__ hce) {
    int g = blockIdx.x * 256 + threadIdx.x;   // 752*256 total
    int m = g >> 8, k = g & 255;              // m = t*16+b
    int t = m >> 4, b = m & 15;
    float v = demb[(long)tgt[b * TT + t] * EE + k];
    dec_e[(long)m * EE + k] = v;
    hce[(long)m * 1280 + 1024 + k] = v;       // e-part of [h|c|e]
}

__global__ void wcs_kernel(const float* __restrict__ attW, float* __restrict__ wcs) {
    int j = blockIdx.x * 256 + threadIdx.x;   // 512 total
    float s = 0.0f;
    for (int k = 2 * HH + EE - HH; k < 2 * HH + EE; ++k)   // cols 768..1279
        s += attW[(long)j * (2 * HH + EE) + k];
    wcs[j] = s;
}

__global__ void zero_t47_kernel(float* __restrict__ out) {
    int g = blockIdx.x * 256 + threadIdx.x;   // 16*32000
    int b = g / VV, v = g % VV;
    out[((long)b * TT + (TT - 1)) * VV + v] = 0.0f;
}

// ---------------------------------------------------------------------------
// Generic WMMA GEMM: C[m][n] = act( sum_k A[m][k]*W[n][k] + bias[n] )
// A rows either direct (A + m*lda) or gathered (gtbl + gidx[m]*lda).
// Block = 256 threads = 8 waves; wave -> one 16x16 N tile; blockIdx.x -> M tile.
// M,N,K all multiples of 16/16/32 in every use here.
// ---------------------------------------------------------------------------
__global__ __launch_bounds__(256)
void wmma_gemm_kernel(const float* __restrict__ A, int lda,
                      const float* __restrict__ gtbl, const int* __restrict__ gidx,
                      const float* __restrict__ W, int ldb, int bOff,
                      float* __restrict__ C, int ldc,
                      const float* __restrict__ bias, int relu, int K) {
    int lane = threadIdx.x & 31, wave = threadIdx.x >> 5;
    int mtile = blockIdx.x;
    int ntile = blockIdx.y * 8 + wave;
    int r = lane & 15, kh = lane >> 4;

    int mrow = mtile * 16 + r;
    const float* arow = gidx ? (gtbl + (long)gidx[mrow] * lda)
                             : (A + (long)mrow * lda);
    const float* brow = W + (long)(ntile * 16 + r) * ldb + bOff;

    v8f c = zero8();
    for (int k0 = 0; k0 < K; k0 += 32) {
        v16h a = frag_f32(arow + k0 + kh * 8);
        v16h b = frag_f32(brow + k0 + kh * 8);
        c = wmma32(a, b, c);
    }

    int ncol = ntile * 16 + (lane & 15);
    float bv = bias ? bias[ncol] : 0.0f;
#pragma unroll
    for (int i = 0; i < 8; ++i) {
        int m = mtile * 16 + i + 8 * kh;
        float v = c[i] + bv;
        if (relu) v = fmaxf(v, 0.0f);
        C[(long)m * ldc + ncol] = v;
    }
}

// ---------------------------------------------------------------------------
// Encoder GRU recurrence. One workgroup, 1024 threads = 32 waves.
// Wave w owns columns [w*16, w*16+16) of all three gates: its 3 WMMA
// accumulators hold gh_r/gh_z/gh_n for those columns, so the gate math is
// done entirely in D-fragment registers. h lives in LDS (f32, 32 KB).
// gi_all (x@Wih.T + bih) is precomputed for all timesteps.
// ---------------------------------------------------------------------------
__global__ __launch_bounds__(1024)
void enc_recurrence_kernel(const float* __restrict__ gi_all,
                           const float* __restrict__ Whh,
                           const float* __restrict__ bhh,
                           const int* __restrict__ src_len,
                           float* __restrict__ outs,   // [B,S,H] masked hn
                           float* __restrict__ hT) {
    __shared__ float hbuf[BB * HH];
    __shared__ int lenl[BB];
    int tid = threadIdx.x, lane = tid & 31, wave = tid >> 5;
    for (int i = tid; i < BB * HH; i += 1024) hbuf[i] = 0.0f;
    if (tid < BB) lenl[tid] = src_len[tid];
    __syncthreads();

    int r = lane & 15, kh = lane >> 4;
    int jt = wave;                       // 0..31 -> column tile within H
    int jcol = jt * 16 + (lane & 15);

    for (int t = 0; t < SS; ++t) {
        v8f acc[3] = {zero8(), zero8(), zero8()};
        for (int k0 = 0; k0 < HH; k0 += 32) {
            v16h a = frag_f32(&hbuf[r * HH + k0 + kh * 8]);
#pragma unroll
            for (int g = 0; g < 3; ++g) {
                v16h b = frag_f32(Whh + (long)(g * HH + jt * 16 + r) * HH + k0 + kh * 8);
                acc[g] = wmma32(a, b, acc[g]);
            }
        }
        float hnew[8];
#pragma unroll
        for (int i = 0; i < 8; ++i) {
            int b = i + 8 * kh;
            const float* gi = gi_all + ((long)t * BB + b) * (3 * HH);
            float rg = sigmoidf_(gi[jcol] + acc[0][i] + bhh[jcol]);
            float zg = sigmoidf_(gi[HH + jcol] + acc[1][i] + bhh[HH + jcol]);
            float hn = acc[2][i] + bhh[2 * HH + jcol];
            float ng = tanhf(gi[2 * HH + jcol] + rg * hn);
            float hold = hbuf[b * HH + jcol];
            float hv = (1.0f - zg) * ng + zg * hold;
            bool valid = (t < lenl[b]);
            hnew[i] = valid ? hv : hold;
            outs[((long)b * SS + t) * HH + jcol] = valid ? hv : 0.0f;
        }
        __syncthreads();
#pragma unroll
        for (int i = 0; i < 8; ++i) {
            int b = i + 8 * kh;
            hbuf[b * HH + jcol] = hnew[i];
        }
        __syncthreads();
    }
    for (int i = tid; i < BB * HH; i += 1024) hT[i] = hbuf[i];
}

// ---------------------------------------------------------------------------
// Decoder: attention + GRU recurrence. One workgroup, 1024 threads = 32 waves.
// Attention is algebraically reduced:
//   score[b,s] = att_vb + sum_j att_v[j]*tanh(pre_enc[b,s,j] + pre_e[t,b,j]
//                                             + cov[b,s]*wcs[j])
// (att_b folded into pre_enc). cov lives in 4 registers per thread.
// Dynamic LDS = h f32 (32K) + scores f32 (16K) + context f16 (16K) = 64 KB.
// ---------------------------------------------------------------------------
__global__ __launch_bounds__(1024)
void dec_recurrence_kernel(const float* __restrict__ pre_enc,   // [B*S,512]
                           const float* __restrict__ pre_e,     // [47*16,512]
                           const float* __restrict__ dec_e,     // [47*16,256]
                           const float* __restrict__ enc_out,   // [B*S,512]
                           const float* __restrict__ attv,
                           const float* __restrict__ wcs,
                           const float* __restrict__ attvb_p,
                           const float* __restrict__ Wih,       // [1536,768]
                           const float* __restrict__ Whh,       // [1536,512]
                           const float* __restrict__ bih,
                           const float* __restrict__ bhh,
                           const int* __restrict__ src_len,
                           const float* __restrict__ hT,
                           float* __restrict__ hce) {            // [47*16,1280]
    extern __shared__ float smem[];
    float* hbuf = smem;                      // 16*512 f32
    float* scores = hbuf + BB * HH;          // 16*256 f32 (scores, then weights)
    _Float16* ecc = (_Float16*)(scores + BB * SS);   // 16*512 f16 (context)

    int tid = threadIdx.x, lane = tid & 31, wave = tid >> 5;
    float attvb = attvb_p[0];
    for (int i = tid; i < BB * HH; i += 1024) hbuf[i] = hT[i];
    float cov[4] = {0.0f, 0.0f, 0.0f, 0.0f};
    __syncthreads();

    int r = lane & 15, kh = lane >> 4;
    int jt = wave, jcol = jt * 16 + (lane & 15);

    for (int t = 0; t < TT - 1; ++t) {
        // ---- phase 1: coverage update + attention scores ----
#pragma unroll
        for (int k = 0; k < 4; ++k) {
            int p = tid + k * 1024;          // 0..4095 -> (b,s)
            int b = p >> 8, s = p & 255;
            if (t > 0) cov[k] += scores[p];  // scores holds prev-step softmax w
            float sc;
            if (s < src_len[b]) {
                const float* pe = pre_enc + (long)p * HH;
                const float* pt = pre_e + ((long)t * BB + b) * HH;
                float cv = cov[k];
                sc = attvb;
                for (int j = 0; j < HH; ++j)
                    sc += attv[j] * tanhf(pe[j] + pt[j] + cv * wcs[j]);
            } else {
                sc = -3.0e38f;
            }
            scores[p] = sc;
        }
        __syncthreads();

        // ---- phase 2: masked softmax, one wave per batch row ----
        if (wave < BB) {
            int b = wave;
            float mx = -3.0e38f;
            for (int s = lane; s < SS; s += 32) mx = fmaxf(mx, scores[b * SS + s]);
            for (int o = 16; o; o >>= 1) mx = fmaxf(mx, __shfl_xor(mx, o, 32));
            float sum = 0.0f;
            for (int s = lane; s < SS; s += 32) {
                float e = __expf(scores[b * SS + s] - mx);
                scores[b * SS + s] = e;
                sum += e;
            }
            for (int o = 16; o; o >>= 1) sum += __shfl_xor(sum, o, 32);
            float inv = 1.0f / sum;
            for (int s = lane; s < SS; s += 32) scores[b * SS + s] *= inv;
        }
        __syncthreads();

        // ---- phase 3: context c[b,h] = sum_s w[b,s]*enc_out[b,s,h] ----
#pragma unroll
        for (int k = 0; k < 8; ++k) {
            int o = tid + k * 1024;          // 0..8191 -> (b,h)
            int b = o >> 9, h = o & 511;
            const float* wr = scores + b * SS;
            const float* eo = enc_out + ((long)b * SS) * HH + h;
            float acc = 0.0f;
            for (int s = 0; s < SS; ++s) acc += wr[s] * eo[(long)s * HH];
            ecc[b * HH + h] = (_Float16)acc;
            hce[((long)t * BB + b) * 1280 + HH + h] = acc;   // c-part
        }
        __syncthreads();

        // ---- phase 4: GRU, x = [e|c] (K=768), h (K=512), 3 gates in regs ----
        v8f agi[3] = {zero8(), zero8(), zero8()};
        v8f agh[3] = {zero8(), zero8(), zero8()};
        for (int k0 = 0; k0 < EE + HH; k0 += 32) {
            v16h a;
            if (k0 < EE)   // e-part from global dec_e (f32 -> f16)
                a = frag_f32(dec_e + ((long)t * BB + r) * EE + k0 + kh * 8);
            else           // c-part from LDS f16
                a = frag_f16(&ecc[r * HH + (k0 - EE) + kh * 8]);
#pragma unroll
            for (int g = 0; g < 3; ++g) {
                v16h b = frag_f32(Wih + (long)(g * HH + jt * 16 + r) * (EE + HH) + k0 + kh * 8);
                agi[g] = wmma32(a, b, agi[g]);
            }
        }
        for (int k0 = 0; k0 < HH; k0 += 32) {
            v16h a = frag_f32(&hbuf[r * HH + k0 + kh * 8]);
#pragma unroll
            for (int g = 0; g < 3; ++g) {
                v16h b = frag_f32(Whh + (long)(g * HH + jt * 16 + r) * HH + k0 + kh * 8);
                agh[g] = wmma32(a, b, agh[g]);
            }
        }
        float hnew[8];
#pragma unroll
        for (int i = 0; i < 8; ++i) {
            int b = i + 8 * kh;
            float rg = sigmoidf_(agi[0][i] + bih[jcol] + agh[0][i] + bhh[jcol]);
            float zg = sigmoidf_(agi[1][i] + bih[HH + jcol] + agh[1][i] + bhh[HH + jcol]);
            float hn = agh[2][i] + bhh[2 * HH + jcol];
            float ng = tanhf(agi[2][i] + bih[2 * HH + jcol] + rg * hn);
            float hold = hbuf[b * HH + jcol];
            hnew[i] = (1.0f - zg) * ng + zg * hold;
        }
        __syncthreads();
#pragma unroll
        for (int i = 0; i < 8; ++i) {
            int b = i + 8 * kh;
            hbuf[b * HH + jcol] = hnew[i];
            hce[((long)t * BB + b) * 1280 + jcol] = hnew[i];   // h-part
        }
        __syncthreads();
    }
}

// ---------------------------------------------------------------------------
// Vocab projection: logits[t,b,v] = hce[t*16+b,:] . out_W[v,:] + out_b[v],
// scattered into d_out as [B, T, V].  grid = (47 Mtiles, 250 groups of 8 Ntiles)
// ---------------------------------------------------------------------------
__global__ __launch_bounds__(256)
void out_gemm_kernel(const float* __restrict__ hce,
                     const float* __restrict__ outW,
                     const float* __restrict__ outb,
                     float* __restrict__ out) {
    int lane = threadIdx.x & 31, wave = threadIdx.x >> 5;
    int mtile = blockIdx.x;                 // 0..46
    int ntile = blockIdx.y * 8 + wave;      // 0..1999
    int r = lane & 15, kh = lane >> 4;

    const float* arow = hce + (long)(mtile * 16 + r) * 1280;
    const float* brow = outW + (long)(ntile * 16 + r) * 1280;

    v8f c = zero8();
    for (int k0 = 0; k0 < 1280; k0 += 32) {
        __builtin_prefetch(brow + k0 + 256, 0, 0);   // global_prefetch_b8
        v16h a = frag_f32(arow + k0 + kh * 8);
        v16h b = frag_f32(brow + k0 + kh * 8);
        c = wmma32(a, b, c);
    }

    int v = ntile * 16 + (lane & 15);
    float bv = outb[v];
#pragma unroll
    for (int i = 0; i < 8; ++i) {
        int m = mtile * 16 + i + 8 * kh;
        int t = m >> 4, b = m & 15;
        out[((long)b * TT + t) * VV + v] = c[i] + bv;
    }
}

// ---------------------------------------------------------------------------
// Host-side orchestration (graph-capture safe: launches only).
// ---------------------------------------------------------------------------
extern "C" void kernel_launch(void* const* d_in, const int* in_sizes, int n_in,
                              void* d_out, int out_size, void* d_ws, size_t ws_size,
                              hipStream_t stream) {
    const int*   src      = (const int*)d_in[0];
    const int*   tgt      = (const int*)d_in[1];
    const int*   src_len  = (const int*)d_in[2];
    const float* enc_emb  = (const float*)d_in[3];
    const float* enc_Wih  = (const float*)d_in[4];
    const float* enc_Whh  = (const float*)d_in[5];
    const float* enc_bih  = (const float*)d_in[6];
    const float* enc_bhh  = (const float*)d_in[7];
    const float* lin_W    = (const float*)d_in[8];
    const float* lin_b    = (const float*)d_in[9];
    const float* att_W    = (const float*)d_in[10];
    const float* att_b    = (const float*)d_in[11];
    const float* att_v    = (const float*)d_in[12];
    const float* att_vb   = (const float*)d_in[13];
    // d_in[14] = dec_emb
    const float* dec_emb  = (const float*)d_in[14];
    const float* dec_Wih  = (const float*)d_in[15];
    const float* dec_Whh  = (const float*)d_in[16];
    const float* dec_bih  = (const float*)d_in[17];
    const float* dec_bhh  = (const float*)d_in[18];
    const float* out_W    = (const float*)d_in[19];
    const float* out_b    = (const float*)d_in[20];
    float* out = (float*)d_out;

    // workspace carve-up (bytes)
    char* ws = (char*)d_ws;
    size_t off = 0;
    auto take = [&](size_t bytes) { char* p = ws + off; off += (bytes + 255) & ~(size_t)255; return p; };
    int*   idx_src  = (int*)take(4096 * 4);
    int*   idx_tgt  = (int*)take(752 * 4);
    float* gi_all   = (float*)take((size_t)4096 * 1536 * 4);   // later reused as pre_enc
    float* enc_outs = (float*)take((size_t)4096 * 512 * 4);
    float* enc_outp = (float*)take((size_t)4096 * 512 * 4);
    float* hT       = (float*)take((size_t)16 * 512 * 4);
    float* dec_e    = (float*)take((size_t)752 * 256 * 4);
    float* pre_e    = (float*)take((size_t)752 * 512 * 4);
    float* wcs      = (float*)take(512 * 4);
    float* hce      = (float*)take((size_t)752 * 1280 * 4);
    float* pre_enc  = gi_all;   // alias: gi_all is dead once the encoder recurrence ends

    // 1) gather row indices for embedding GEMMs
    build_idx_kernel<<<16, 256, 0, stream>>>(src, tgt, idx_src, idx_tgt);

    // 2) gi_all[t*16+b, 3H] = enc_emb[src] @ enc_Wih.T + enc_bih   (M=4096,K=256,N=1536)
    wmma_gemm_kernel<<<dim3(256, 12), 256, 0, stream>>>(
        nullptr, 256, enc_emb, idx_src, enc_Wih, 256, 0,
        gi_all, 1536, enc_bih, 0, 256);

    // 3) encoder GRU recurrence (sequential over S=256)
    enc_recurrence_kernel<<<1, 1024, 0, stream>>>(
        gi_all, enc_Whh, enc_bhh, src_len, enc_outs, hT);

    // 4) enc_output = relu(outs @ lin_W.T + lin_b)   (M=4096,K=512,N=512)
    wmma_gemm_kernel<<<dim3(256, 4), 256, 0, stream>>>(
        enc_outs, 512, nullptr, nullptr, lin_W, 512, 0,
        enc_outp, 512, lin_b, 1, 512);

    // 5) pre_enc = enc_output @ W_enc.T + att_b  (att_W cols 0:512)
    wmma_gemm_kernel<<<dim3(256, 4), 256, 0, stream>>>(
        enc_outp, 512, nullptr, nullptr, att_W, 1280, 0,
        pre_enc, 512, att_b, 0, 512);

    // 6) decoder embeddings: dec_e + e-part of hce
    dec_embed_kernel<<<752, 256, 0, stream>>>(tgt, dec_emb, dec_e, hce);

    // 7) pre_e = dec_e @ W_e.T (att_W cols 512:768), M=752,K=256,N=512
    wmma_gemm_kernel<<<dim3(47, 4), 256, 0, stream>>>(
        nullptr, 256, dec_emb, idx_tgt, att_W, 1280, 512,
        pre_e, 512, nullptr, 0, 256);

    // 8) wcs[j] = rowsum of W_cov (att_W cols 768:1280)
    wcs_kernel<<<2, 256, 0, stream>>>(att_W, wcs);

    // 9) decoder recurrence (sequential over T-1=47), 64 KB dynamic LDS
    size_t dec_lds = (size_t)(16 * 512 + 16 * 256) * 4 + (size_t)(16 * 512) * 2;
    dec_recurrence_kernel<<<1, 1024, dec_lds, stream>>>(
        pre_enc, pre_e, dec_e, enc_outp, att_v, wcs, att_vb,
        dec_Wih, dec_Whh, dec_bih, dec_bhh, src_len, hT, hce);

    // 10) zero-fill logits at t = T-1
    zero_t47_kernel<<<2000, 256, 0, stream>>>(out);

    // 11) vocab projection GEMM: [752,1280] x [32000,1280]^T -> out[B,T,V]
    out_gemm_kernel<<<dim3(47, 250), 256, 0, stream>>>(hce, out_W, out_b, out);

    (void)in_sizes; (void)n_in; (void)out_size; (void)ws_size;
}
